// TinyCausalTransformer_78675210928142
// MI455X (gfx1250) — compile-verified, matrix operand order checked by reading, not run on hardware
//
#include <hip/hip_runtime.h>
#include <hip/hip_bf16.h>

// ---------------------------------------------------------------------------
// Types & helpers for CDNA5 WMMA (gfx1250, wave32)
// ---------------------------------------------------------------------------
typedef __attribute__((ext_vector_type(16))) __bf16 v16bf;
typedef __attribute__((ext_vector_type(8)))  float  v8f;

__device__ __forceinline__ __bf16 f2bf(float x) {
    return static_cast<__bf16>(x);            // let backend use native cvt if present
}

// Pack two fp32 -> packed bf16 dword for wide LDS stores.
__device__ __forceinline__ unsigned pack_bf16(float a, float b) {
    union { __bf16 h[2]; unsigned u; } p;
    p.h[0] = static_cast<__bf16>(a);
    p.h[1] = static_cast<__bf16>(b);
    return p.u;
}

// A/B fragment loader for V_WMMA_*_16X16X32_BF16 from a bf16 LDS tile.
// lane = hf*16 + r ; lane holds row r, K = {hf*8..hf*8+7, 16+hf*8..16+hf*8+7}
__device__ __forceinline__ v16bf load_frag(const __bf16* base, int stride, int r, int hf) {
    const __bf16* p = base + r * stride;
    v16bf f;
#pragma unroll
    for (int i = 0; i < 8; ++i) {
        f[i]     = p[hf * 8 + i];
        f[8 + i] = p[16 + hf * 8 + i];
    }
    return f;
}

__device__ __forceinline__ v8f wmma_bf16(v16bf a, v16bf b, v8f c) {
    return __builtin_amdgcn_wmma_f32_16x16x32_bf16(false, a, false, b, (short)0, c, false, false);
}

// ---------------------------------------------------------------------------
// Generic tiled WMMA GEMM:  C[M,N] (+)= act(A[M,K] @ W[N,K]^T + bias[N])
//   EPI: 0 = store fp32, 1 = add into C (residual), 2 = weighted scatter-add,
//        3 = store bf16 row-major (C is __bf16[M][ldc]; QK projection)
//        4 = store bf16 TRANSPOSED (C is __bf16[N][ldc]; V projection ->
//            vt[dim][token]; 8 consecutive tokens/lane = contiguous stores)
//   GATHER_A: A row m is A[row_idx[m]]
//   GELU: apply exact gelu to the epilogue value
// Block = 256 threads (8 waves). Block tile 128x64, K-step 32.
// ---------------------------------------------------------------------------
#define TILE_M 128
#define TILE_N 64
#define TILE_K 32
#define LDT    40   // padded LDS row stride (bf16) -> conflict-free banks

template<int EPI, int GATHER_A, int GELU>
__global__ __launch_bounds__(256)
void gemm_kernel(const float* __restrict__ A, const float* __restrict__ W,
                 const float* __restrict__ bias, float* __restrict__ C,
                 int M, int N, int K, int ldc,
                 const int* __restrict__ row_idx, const float* __restrict__ row_w,
                 const int* __restrict__ count_ptr)
{
    if (count_ptr) M = *count_ptr;            // dynamic row count (MoE capacity grid)
    const int bm = blockIdx.x * TILE_M;
    const int bn = blockIdx.y * TILE_N;
    if (bm >= M) return;

    __shared__ __bf16 sA[TILE_M * LDT];
    __shared__ __bf16 sB[TILE_N * LDT];

    const int tid  = threadIdx.x;
    const int wave = tid >> 5;
    const int lane = tid & 31;
    const int hf   = (lane >> 4) & 1;
    const int r    = lane & 15;

    v8f acc[4] = {};                          // wave rows [bm+wave*16,+16) x 4 N-subtiles

    for (int k0 = 0; k0 < K; k0 += TILE_K) {
        // stage 192 rows x 32 cols fp32 -> bf16 (packed dword LDS stores)
#pragma unroll
        for (int it = 0; it < 3; ++it) {
            int cid = tid + 256 * it;         // 0..767
            int row = cid >> 2;               // 0..191
            int col = (cid & 3) * 8;
            if (row < TILE_M) {
                int gr = bm + row;
                unsigned* dst = (unsigned*)(sA + row * LDT + col);
                if (gr < M) {
                    int arow = GATHER_A ? row_idx[gr] : gr;
                    const float* src = A + (size_t)arow * K + k0 + col;
                    __builtin_prefetch(src + TILE_K, 0, 0);
#pragma unroll
                    for (int i = 0; i < 4; ++i) dst[i] = pack_bf16(src[2*i], src[2*i+1]);
                } else {
#pragma unroll
                    for (int i = 0; i < 4; ++i) dst[i] = 0u;
                }
            } else {
                int nrow = bn + (row - TILE_M);
                const float* src = W + (size_t)nrow * K + k0 + col;
                __builtin_prefetch(src + TILE_K, 0, 0);
                unsigned* dst = (unsigned*)(sB + (row - TILE_M) * LDT + col);
#pragma unroll
                for (int i = 0; i < 4; ++i) dst[i] = pack_bf16(src[2*i], src[2*i+1]);
            }
        }
        __syncthreads();

        v16bf a = load_frag(sA + (wave * 16) * LDT, LDT, r, hf);
#pragma unroll
        for (int j = 0; j < 4; ++j) {
            v16bf b = load_frag(sB + (j * 16) * LDT, LDT, r, hf);
            acc[j] = wmma_bf16(a, b, acc[j]);
        }
        __syncthreads();
    }

    // epilogue: lane holds col n = bn+j*16+r, rows m = bm+wave*16+hf*8+v
#pragma unroll
    for (int j = 0; j < 4; ++j) {
        int n = bn + j * 16 + r;
        float bv = bias ? bias[n] : 0.0f;
#pragma unroll
        for (int v = 0; v < 8; ++v) {
            int m = bm + wave * 16 + hf * 8 + v;
            if (m < M) {
                float val = acc[j][v] + bv;
                if (GELU) val = 0.5f * val * (1.0f + erff(val * 0.70710678118f));
                if (EPI == 0) {
                    C[(size_t)m * ldc + n] = val;
                } else if (EPI == 1) {
                    C[(size_t)m * ldc + n] += val;
                } else if (EPI == 2) {
                    int tok = row_idx[m];
                    atomicAdd(&C[(size_t)tok * ldc + n], row_w[m] * val);
                } else if (EPI == 3) {
                    ((__bf16*)C)[(size_t)m * ldc + n] = f2bf(val);
                } else {
                    ((__bf16*)C)[(size_t)n * ldc + m] = f2bf(val);  // transposed
                }
            }
        }
    }
}

// ---------------------------------------------------------------------------
// Flash attention, one wave per (16-query tile, head). head_dim = 64.
// qk: bf16 [tok][1536] = [Q(768) | K(768)]  (EPI=3 GEMM output)
// vt: bf16 [768 dims][VLD tokens]           (EPI=4 GEMM output, transposed)
// Both buffers have full 2048-row/col capacity, so unguarded async over-reads
// stay inside the allocations and are masked out of softmax (P==0 exactly).
// K/V tiles staged with GLOBAL_LOAD_ASYNC_TO_LDS_B128 (ASYNCcnt path);
// bf16 end-to-end, all fragment feeds are wide ds_load_b128.
// LDS row strides padded for conflict-free banks (72*2B=36, 40*2B=20 banks).
// ---------------------------------------------------------------------------
#define QKLD  1536   // qk row stride (elements)
#define VLD   2048   // vt row capacity (tokens)
#define KSTR  72     // LDS stride for Q/K tiles
#define VTSTR 40     // LDS stride for Vt tile
#define PSTR  40     // LDS stride for P tile

__global__ __launch_bounds__(32)
void attn_kernel(const __bf16* __restrict__ qk, const __bf16* __restrict__ vt,
                 float* __restrict__ out, int T, int D)
{
    const int HD = 64;
    const int h  = blockIdx.y;
    const int qb = blockIdx.x * 16;
    if (qb >= T) return;
    const __bf16* qp = qk + h * HD;
    const __bf16* kp = qk + D + h * HD;
    const __bf16* vp = vt + (size_t)(h * HD) * VLD;   // rows = dims, cols = tokens

    __shared__ __bf16 sQ[16 * KSTR];
    __shared__ __bf16 fK[32 * KSTR];    // K rows (keys x dims), row reads contiguous
    __shared__ __bf16 fVt[64 * VTSTR];  // V^T rows (dims x keys), row reads contiguous
    __shared__ __bf16 sP[16 * PSTR];

    const int lane = threadIdx.x;
    const int hf = lane >> 4;
    const int r  = lane & 15;

    const unsigned kl_base = (unsigned)(uintptr_t)fK;   // LDS byte offset (addr[31:0])
    const unsigned vl_base = (unsigned)(uintptr_t)fVt;

    // stage Q tile (16 x 64) once
    for (int idx = lane; idx < 16 * 64; idx += 32) {
        int row = idx >> 6, d = idx & 63;
        int q = qb + row;
        sQ[row * KSTR + d] = (q < T) ? qp[(size_t)q * QKLD + d] : (__bf16)0.0f;
    }
    __syncthreads();
    v16bf aq0 = load_frag(sQ, KSTR, r, hf);
    v16bf aq1 = load_frag(sQ + 32, KSTR, r, hf);

    v8f accO[4] = {};
    float mrow[8], lrow[8];
#pragma unroll
    for (int v = 0; v < 8; ++v) { mrow[v] = -1e30f; lrow[v] = 0.0f; }

    const int kend = qb + 16;                 // causal: keys <= last query of tile
    for (int kb = 0; kb < kend && kb < T; kb += 32) {
        // ---- async stage K (32 keys x 64 dims) and Vt (64 dims x 32 keys) ----
        asm volatile("s_wait_dscnt 0x0" ::: "memory");   // prior LDS reads done
#pragma unroll
        for (int it = 0; it < 8; ++it) {
            int c  = lane + 32 * it;          // 0..255
            // K: 32 rows x 8 chunks (16B = 8 bf16 dims)
            int kk = c >> 3;
            int cb = c & 7;
            unsigned long long gak =
                (unsigned long long)(uintptr_t)(kp + (size_t)(kb + kk) * QKLD) + (unsigned)(cb * 16);
            unsigned lak = kl_base + (unsigned)(kk * (KSTR * 2) + cb * 16);
            asm volatile("global_load_async_to_lds_b128 %0, %1, off"
                         :: "v"(lak), "v"(gak) : "memory");
            // Vt: 64 rows x 4 chunks (16B = 8 bf16 keys)
            int d  = c >> 2;
            int vb = c & 3;
            unsigned long long gav =
                (unsigned long long)(uintptr_t)(vp + (size_t)d * VLD + kb) + (unsigned)(vb * 16);
            unsigned lav = vl_base + (unsigned)(d * (VTSTR * 2) + vb * 16);
            asm volatile("global_load_async_to_lds_b128 %0, %1, off"
                         :: "v"(lav), "v"(gav) : "memory");
        }
        asm volatile("s_wait_asynccnt 0x0" ::: "memory");
        __syncthreads();

        // ---- S = Q K^T (16 x 32) ----
        v8f accS[2] = {};
#pragma unroll
        for (int j = 0; j < 2; ++j) {
            v16bf bk0 = load_frag(fK + (j * 16) * KSTR, KSTR, r, hf);
            v16bf bk1 = load_frag(fK + (j * 16) * KSTR + 32, KSTR, r, hf);
            accS[j] = wmma_bf16(aq0, bk0, accS[j]);
            accS[j] = wmma_bf16(aq1, bk1, accS[j]);
        }

        // ---- scale + causal mask + online softmax update ----
        float s[2][8];
#pragma unroll
        for (int j = 0; j < 2; ++j)
#pragma unroll
            for (int v = 0; v < 8; ++v) {
                int qg = qb + hf * 8 + v;
                int kg = kb + j * 16 + r;
                float x = accS[j][v] * 0.125f;       // 1/sqrt(64)
                if (kg > qg || kg >= T) x = -1e30f;
                s[j][v] = x;
            }
#pragma unroll
        for (int v = 0; v < 8; ++v) {
            float mx = fmaxf(s[0][v], s[1][v]);
#pragma unroll
            for (int off = 1; off < 16; off <<= 1)
                mx = fmaxf(mx, __shfl_xor(mx, off, 16));
            float mnew  = fmaxf(mrow[v], mx);
            float alpha = __expf(mrow[v] - mnew);
            float p0 = (s[0][v] <= -1e29f) ? 0.0f : __expf(s[0][v] - mnew);
            float p1 = (s[1][v] <= -1e29f) ? 0.0f : __expf(s[1][v] - mnew);
            float psum = p0 + p1;
#pragma unroll
            for (int off = 1; off < 16; off <<= 1)
                psum += __shfl_xor(psum, off, 16);
            lrow[v] = lrow[v] * alpha + psum;
            mrow[v] = mnew;
#pragma unroll
            for (int j2 = 0; j2 < 4; ++j2) accO[j2][v] *= alpha;
            sP[(hf * 8 + v) * PSTR + r]      = f2bf(p0);
            sP[(hf * 8 + v) * PSTR + 16 + r] = f2bf(p1);
        }
        __syncthreads();

        // ---- O += P V : P is 16x32 A-operand; Vt rows give B-operand ----
        v16bf pf = load_frag(sP, PSTR, r, hf);
#pragma unroll
        for (int j2 = 0; j2 < 4; ++j2) {
            v16bf bv = load_frag(fVt + (j2 * 16) * VTSTR, VTSTR, r, hf);
            accO[j2] = wmma_bf16(pf, bv, accO[j2]);
        }
        __syncthreads();
    }

    // normalize and write (fp32)
#pragma unroll
    for (int j2 = 0; j2 < 4; ++j2) {
        int d = h * 64 + j2 * 16 + r;
#pragma unroll
        for (int v = 0; v < 8; ++v) {
            int q = qb + hf * 8 + v;
            if (q < T) {
                float li = lrow[v];
                out[(size_t)q * D + d] = (li > 0.0f) ? accO[j2][v] / li : 0.0f;
            }
        }
    }
}

// ---------------------------------------------------------------------------
// LayerNorm: one block per row
// ---------------------------------------------------------------------------
__global__ __launch_bounds__(256)
void ln_kernel(const float* __restrict__ x, const float* __restrict__ s,
               const float* __restrict__ b, float* __restrict__ o, int D)
{
    const int row = blockIdx.x;
    const float* xr = x + (size_t)row * D;
    float* orow = o + (size_t)row * D;
    float ls = 0.0f, ls2 = 0.0f;
    for (int i = threadIdx.x; i < D; i += 256) { float v = xr[i]; ls += v; ls2 += v * v; }
    __shared__ float sb[16];
    __shared__ float mv[2];
    int lane = threadIdx.x & 31, w = threadIdx.x >> 5;
    for (int off = 16; off; off >>= 1) {
        ls  += __shfl_xor(ls, off, 32);
        ls2 += __shfl_xor(ls2, off, 32);
    }
    if (lane == 0) { sb[w] = ls; sb[8 + w] = ls2; }
    __syncthreads();
    if (threadIdx.x == 0) {
        float a = 0, c = 0;
        for (int i = 0; i < 8; ++i) { a += sb[i]; c += sb[8 + i]; }
        float mean = a / D;
        float var  = c / D - mean * mean;
        mv[0] = mean; mv[1] = rsqrtf(var + 1e-5f);
    }
    __syncthreads();
    float mean = mv[0], rstd = mv[1];
    for (int i = threadIdx.x; i < D; i += 256)
        orow[i] = (xr[i] - mean) * rstd * s[i] + b[i];
}

// ---------------------------------------------------------------------------
// MoE router / top-2 / bookkeeping
// ---------------------------------------------------------------------------
__global__ __launch_bounds__(32)
void router_kernel(const float* __restrict__ h, const float* __restrict__ rw,
                   const float* __restrict__ extra, float* __restrict__ logits, int D)
{
    const int t = blockIdx.x;
    const float* hr = h + (size_t)t * D;
    const int lane = threadIdx.x;
    for (int e = 0; e < 8; ++e) {
        float sum = 0.0f;
        for (int i = lane; i < D; i += 32) sum += hr[i] * rw[e * D + i];
        for (int off = 16; off; off >>= 1) sum += __shfl_xor(sum, off, 32);
        if (lane == 0) logits[t * 8 + e] = sum + (extra ? extra[e] : 0.0f);
    }
}

__global__ void top2_kernel(const float* __restrict__ logits, int M, int cap,
                            int* __restrict__ counts, int* __restrict__ list,
                            float* __restrict__ wlist,
                            float* __restrict__ probsum, float* __restrict__ selcnt)
{
    int t = blockIdx.x * blockDim.x + threadIdx.x;
    if (t >= M) return;
    float p[8]; float mx = -1e30f;
    for (int e = 0; e < 8; ++e) { p[e] = logits[t * 8 + e]; mx = fmaxf(mx, p[e]); }
    float sum = 0.0f;
    for (int e = 0; e < 8; ++e) { p[e] = __expf(p[e] - mx); sum += p[e]; }
    float inv = 1.0f / sum;
    for (int e = 0; e < 8; ++e) p[e] *= inv;
    int i1 = 0;
    for (int e = 1; e < 8; ++e) if (p[e] > p[i1]) i1 = e;
    int i2 = (i1 == 0) ? 1 : 0;
    for (int e = 0; e < 8; ++e) if (e != i1 && p[e] > p[i2]) i2 = e;
    float wn = 1.0f / (p[i1] + p[i2]);
    for (int e = 0; e < 8; ++e) atomicAdd(&probsum[e], p[e]);
    atomicAdd(&selcnt[i1], 1.0f);
    atomicAdd(&selcnt[i2], 1.0f);
    int s1 = atomicAdd(&counts[i1], 1); list[i1 * cap + s1] = t; wlist[i1 * cap + s1] = p[i1] * wn;
    int s2 = atomicAdd(&counts[i2], 1); list[i2 * cap + s2] = t; wlist[i2 * cap + s2] = p[i2] * wn;
}

__global__ void zero_moe(int* counts, float* probsum, float* selcnt) {
    int i = threadIdx.x;
    if (i < 8) { counts[i] = 0; probsum[i] = 0.0f; selcnt[i] = 0.0f; }
}

__global__ void bal_kernel(const float* probsum, const float* selcnt, float invM, float* bal) {
    if (threadIdx.x == 0) {
        float s = 0.0f;
        for (int e = 0; e < 8; ++e) s += (probsum[e] * invM) * (selcnt[e] * invM);
        *bal += 8.0f * s;
    }
}

__global__ void zero1_kernel(float* p) { if (threadIdx.x == 0) *p = 0.0f; }

// ---------------------------------------------------------------------------
// Elementwise glue kernels
// ---------------------------------------------------------------------------
__global__ void add_pos_kernel(const float* a, const float* b, float* o, int n) {
    int i = blockIdx.x * 256 + threadIdx.x;
    if (i < n) o[i] = a[i] + b[i];
}

__global__ void build_gin_kernel(const float* __restrict__ lat, const float* __restrict__ hloc,
                                 float* __restrict__ gin, int D, int CH, int S)
{
    int i = blockIdx.x * 256 + threadIdx.x;
    if (i >= S * D) return;
    int row = i / D, d = i - row * D;
    gin[i] = (row == 0) ? lat[d] : hloc[(size_t)((row - 1) * CH + CH - 1) * D + d];
}

__global__ void combine_kernel(const float* __restrict__ hloc, const float* __restrict__ hg,
                               float* __restrict__ o, int T, int D, int CH)
{
    int i = blockIdx.x * 256 + threadIdx.x;
    if (i >= T * D) return;
    int t = i / D, d = i - t * D;
    o[i] = hloc[i] + hg[(size_t)(t / CH) * D + d];
}

__global__ __launch_bounds__(32)
void latrouter_kernel(const float* __restrict__ lat, const float* __restrict__ lrw,
                      float* __restrict__ out, int D)
{
    const int lane = threadIdx.x;
    for (int e = 0; e < 8; ++e) {
        float s = 0.0f;
        for (int i = lane; i < D; i += 32) s += lat[i] * lrw[e * D + i];
        for (int off = 16; off; off >>= 1) s += __shfl_xor(s, off, 32);
        if (lane == 0) out[e] = s;
    }
}

__global__ void store_bal_kernel(const float* bal, float* dst) {
    if (threadIdx.x == 0) dst[0] = *bal;
}

// ---------------------------------------------------------------------------
// Host orchestration
// ---------------------------------------------------------------------------
extern "C" void kernel_launch(void* const* d_in, const int* in_sizes, int n_in,
                              void* d_out, int out_size, void* d_ws, size_t ws_size,
                              hipStream_t stream)
{
    (void)in_sizes; (void)n_in; (void)out_size; (void)ws_size;
    const int T = 2048, D = 768, D3 = 2304, D4 = 3072, Hh = 12, CH = 16;
    const int S = T / CH + 1;          // 129 global tokens
    const int VOCAB = 32000, CAP = 2048;

    const float* x_emb  = (const float*)d_in[0];
    const float* latent = (const float*)d_in[1];
    const float* pos    = (const float*)d_in[2];
    const float* l_ln1s = (const float*)d_in[3];
    const float* l_ln1b = (const float*)d_in[4];
    const float* l_ln2s = (const float*)d_in[5];
    const float* l_ln2b = (const float*)d_in[6];
    const float* l_aw   = (const float*)d_in[7];
    const float* l_ab   = (const float*)d_in[8];
    const float* l_ow   = (const float*)d_in[9];
    const float* l_ob   = (const float*)d_in[10];
    const float* l_rw   = (const float*)d_in[11];
    const float* l_w1   = (const float*)d_in[12];
    const float* l_b1   = (const float*)d_in[13];
    const float* l_w2   = (const float*)d_in[14];
    const float* l_b2   = (const float*)d_in[15];
    const float* g_ln1s = (const float*)d_in[16];
    const float* g_ln1b = (const float*)d_in[17];
    const float* g_ln2s = (const float*)d_in[18];
    const float* g_ln2b = (const float*)d_in[19];
    const float* g_aw   = (const float*)d_in[20];
    const float* g_ab   = (const float*)d_in[21];
    const float* g_ow   = (const float*)d_in[22];
    const float* g_ob   = (const float*)d_in[23];
    const float* g_rw   = (const float*)d_in[24];
    const float* g_w1   = (const float*)d_in[25];
    const float* g_b1   = (const float*)d_in[26];
    const float* g_w2   = (const float*)d_in[27];
    const float* g_b2   = (const float*)d_in[28];
    const float* g_lrw  = (const float*)d_in[29];
    const float* ln_s   = (const float*)d_in[30];
    const float* ln_b   = (const float*)d_in[31];
    const float* head_w = (const float*)d_in[32];
    float* out = (float*)d_out;

    // workspace partition (floats)
    float* ws = (float*)d_ws;
    size_t off = 0;
    auto alloc = [&](size_t n) { float* p = ws + off; off += n; return p; };
    float* x      = alloc((size_t)T * D);          // residual stream / h_local
    float* hbuf   = alloc((size_t)T * D);          // LN outputs
    float* qkbuf  = alloc((size_t)T * 2 * D / 2);  // bf16 [2048][1536]
    float* vtbuf  = alloc((size_t)D * 2048 / 2);   // bf16 [768][2048] (transposed V)
    float* obuf   = alloc((size_t)T * D);          // attention output
    float* logits = alloc((size_t)T * 8);
    float* h1     = alloc((size_t)T * D4);         // expert hidden (reused per expert)
    float* gin    = alloc((size_t)S * D);          // global token stream
    float* ocomb  = alloc((size_t)T * D);
    float* probsum = alloc(8);
    float* selcnt  = alloc(8);
    float* balp    = alloc(1);
    float* latlog  = alloc(8);
    float* wlist   = alloc((size_t)8 * CAP);
    int*   counts  = (int*)alloc(8);
    int*   elist   = (int*)alloc((size_t)8 * CAP);

    auto run_moe = [&](const float* hin, const float* rw, const float* w1, const float* b1,
                       const float* w2, const float* b2, const float* extra,
                       float* xres, int M) {
        zero_moe<<<1, 32, 0, stream>>>(counts, probsum, selcnt);
        router_kernel<<<M, 32, 0, stream>>>(hin, rw, extra, logits, D);
        top2_kernel<<<(M + 255) / 256, 256, 0, stream>>>(logits, M, CAP, counts, elist,
                                                          wlist, probsum, selcnt);
        bal_kernel<<<1, 1, 0, stream>>>(probsum, selcnt, 1.0f / (float)M, balp);
        dim3 g1((M + TILE_M - 1) / TILE_M, D4 / TILE_N);
        dim3 g2((M + TILE_M - 1) / TILE_M, D / TILE_N);
        for (int e = 0; e < 8; ++e) {
            // h1 = gelu(gather(hin) @ W1[e]^T + b1[e])
            gemm_kernel<0, 1, 1><<<g1, 256, 0, stream>>>(
                hin, w1 + (size_t)e * D4 * D, b1 + (size_t)e * D4, h1,
                M, D4, D, D4, elist + e * CAP, nullptr, counts + e);
            // x[tok] += w * (h1 @ W2[e]^T + b2[e])
            gemm_kernel<2, 0, 0><<<g2, 256, 0, stream>>>(
                h1, w2 + (size_t)e * D * D4, b2 + (size_t)e * D, xres,
                M, D, D4, D, elist + e * CAP, wlist + e * CAP, counts + e);
        }
    };

    // attention block: QK projection (row-major bf16), V projection (transposed
    // bf16), flash attention, out-projection with fused residual add.
    auto run_attn = [&](const float* hin, const float* aw, const float* ab,
                        const float* ow, const float* ob, float* xres, int M) {
        dim3 gm((M + TILE_M - 1) / TILE_M, 1);
        gm.y = (2 * D) / TILE_N;
        gemm_kernel<3, 0, 0><<<gm, 256, 0, stream>>>(
            hin, aw, ab, qkbuf, M, 2 * D, D, QKLD, nullptr, nullptr, nullptr);
        gm.y = D / TILE_N;
        gemm_kernel<4, 0, 0><<<gm, 256, 0, stream>>>(
            hin, aw + (size_t)2 * D * D, ab + 2 * D, vtbuf,
            M, D, D, VLD, nullptr, nullptr, nullptr);
        attn_kernel<<<dim3((M + 15) / 16, Hh), 32, 0, stream>>>(
            (const __bf16*)qkbuf, (const __bf16*)vtbuf, obuf, M, D);
        gemm_kernel<1, 0, 0><<<dim3((M + TILE_M - 1) / TILE_M, D / TILE_N), 256, 0, stream>>>(
            obuf, ow, ob, xres, M, D, D, D, nullptr, nullptr, nullptr);
    };

    // x = x_emb + pos_emb ; bal = 0
    add_pos_kernel<<<(T * D + 255) / 256, 256, 0, stream>>>(x_emb, pos, x, T * D);
    zero1_kernel<<<1, 1, 0, stream>>>(balp);

    // ---- local layers ----
    for (int l = 0; l < 2; ++l) {
        ln_kernel<<<T, 256, 0, stream>>>(x, l_ln1s + l * D, l_ln1b + l * D, hbuf, D);
        run_attn(hbuf, l_aw + (size_t)l * D3 * D, l_ab + (size_t)l * D3,
                 l_ow + (size_t)l * D * D, l_ob + (size_t)l * D, x, T);
        ln_kernel<<<T, 256, 0, stream>>>(x, l_ln2s + l * D, l_ln2b + l * D, hbuf, D);
        run_moe(hbuf, l_rw + (size_t)l * 8 * D,
                l_w1 + (size_t)l * 8 * D4 * D, l_b1 + (size_t)l * 8 * D4,
                l_w2 + (size_t)l * 8 * D * D4, l_b2 + (size_t)l * 8 * D,
                nullptr, x, T);
    }

    // ---- global layer on [latent | chunk summaries] ----
    latrouter_kernel<<<1, 32, 0, stream>>>(latent, g_lrw, latlog, D);
    build_gin_kernel<<<(S * D + 255) / 256, 256, 0, stream>>>(latent, x, gin, D, CH, S);

    ln_kernel<<<S, 256, 0, stream>>>(gin, g_ln1s, g_ln1b, hbuf, D);
    run_attn(hbuf, g_aw, g_ab, g_ow, g_ob, gin, S);
    ln_kernel<<<S, 256, 0, stream>>>(gin, g_ln2s, g_ln2b, hbuf, D);
    run_moe(hbuf, g_rw, g_w1, g_b1, g_w2, g_b2, latlog, gin, S);

    // ---- combine, final LN, head projection ----
    combine_kernel<<<(T * D + 255) / 256, 256, 0, stream>>>(x, gin, ocomb, T, D, CH);
    ln_kernel<<<T, 256, 0, stream>>>(ocomb, ln_s, ln_b, hbuf, D);
    gemm_kernel<0, 0, 0><<<dim3(T / TILE_M, VOCAB / TILE_N), 256, 0, stream>>>(
        hbuf, head_w, nullptr, out, T, VOCAB, D, VOCAB, nullptr, nullptr, nullptr);
    store_bal_kernel<<<1, 1, 0, stream>>>(balp, out + (size_t)T * VOCAB);
}
